// WindowbasedSelfAttention_53747220742723
// MI455X (gfx1250) — matrix operand order repaired and checked by reading, not compiled, not run on hardware
//
#include <hip/hip_runtime.h>

typedef __attribute__((ext_vector_type(16))) __bf16 v16bf;
typedef __attribute__((ext_vector_type(8)))  __bf16 v8bf;
typedef __attribute__((ext_vector_type(4)))  __bf16 v4bf;
typedef __attribute__((ext_vector_type(8)))  float  v8f;

#define SCALE_ 0.25f   // HD=16 -> 16^-0.5

union BF16x16 { v16bf v; v8bf h[2]; };

static __device__ __forceinline__ v8f wmma_bf16(v16bf a, v16bf b, v8f c) {
  // D = A(16x32 bf16) * B(32x16 bf16) + C(16x16 f32)
  return __builtin_amdgcn_wmma_f32_16x16x32_bf16(false, a, false, b, (short)0, c, false, false);
}

// -------- Weight pre-swizzle: f32 [128 x ncols] -> bf16 B-fragments --------
// Layout: [nt][kt][lane(32)][e(16)]; lane L: n = nt*16 + (L&15),
// element e: k = kt*32 + e + 16*(L>>4)  (contiguous k order, same as the A loads).
__global__ void wconv_kernel(const float* __restrict__ w, __bf16* __restrict__ dst,
                             int ncols, int total) {
  int tid = blockIdx.x * 256 + threadIdx.x;
  if (tid >= total) return;
  int e  = tid & 15;
  int L  = (tid >> 4) & 31;
  int kt = (tid >> 9) & 3;
  int nt = tid >> 11;
  int n = nt * 16 + (L & 15);
  int k = kt * 32 + e + 16 * (L >> 4);
  dst[tid] = (__bf16)w[k * ncols + n];
}

// -------- Fused window attention: one block per window, 8 waves = 8 heads ----
__global__ __launch_bounds__(256)
void swin_attn_kernel(const float* __restrict__ x,
                      const float* __restrict__ gamma,
                      const float* __restrict__ beta,
                      const __bf16* __restrict__ wqkv_bf,  // [24][4][32][16]
                      const float* __restrict__ bqkv,
                      const __bf16* __restrict__ wout_bf,  // [8][4][32][16]
                      const float* __restrict__ bout,
                      float* __restrict__ out) {
  __shared__ __bf16 s_xn[64 * 128];     // xn (bf16), later reused as attn
  __shared__ __bf16 s_q [8][64 * 16];   // q per head, later reused as probs slab
  __shared__ __bf16 s_k [8][64 * 16];   // k per head
  __shared__ __bf16 s_vT[8][16 * 64];   // v transposed per head: [d][token]

  const int tid  = threadIdx.x;
  const int lane = tid & 31;
  const int wv   = tid >> 5;     // wave id == head id
  const int h    = lane >> 4;    // lane half
  const int ln   = lane & 15;

  const int wid = blockIdx.x;    // window id: 0..4095
  const int b   = wid >> 6;
  const int wr  = (wid >> 3) & 7;
  const int wc  = wid & 7;

  // ---------------- Phase 1: load x + LayerNorm -> s_xn (bf16) --------------
  {
    float4 g4 = ((const float4*)gamma)[lane];
    float4 b4 = ((const float4*)beta)[lane];
    for (int t = wv; t < 64; t += 8) {
      if (t < 49) {
        int trow = t / 7, tcol = t - trow * 7;
        int grow = wr * 7 + trow, gcol = wc * 7 + tcol;
        const float4* xp =
            (const float4*)(x + ((size_t)((b * 56 + grow) * 56 + gcol)) * 128);
        float4 xv = xp[lane];
        float a0 = xv.x, a1 = xv.y, a2 = xv.z, a3 = xv.w;
        float s  = a0 + a1 + a2 + a3;
        float s2 = a0 * a0 + a1 * a1 + a2 * a2 + a3 * a3;
#pragma unroll
        for (int o = 1; o < 32; o <<= 1) {
          s  += __shfl_xor(s, o, 32);
          s2 += __shfl_xor(s2, o, 32);
        }
        float mu  = s * (1.0f / 128.0f);
        float var = s2 * (1.0f / 128.0f) - mu * mu;
        float inv = rsqrtf(var + 1e-5f);
        v4bf r;
        r[0] = (__bf16)((a0 - mu) * inv * g4.x + b4.x);
        r[1] = (__bf16)((a1 - mu) * inv * g4.y + b4.y);
        r[2] = (__bf16)((a2 - mu) * inv * g4.z + b4.z);
        r[3] = (__bf16)((a3 - mu) * inv * g4.w + b4.w);
        *(v4bf*)&s_xn[t * 128 + lane * 4] = r;
      } else {
        v4bf z = {};
        *(v4bf*)&s_xn[t * 128 + lane * 4] = z;
      }
    }
  }
  __syncthreads();

  // ---------------- Phase 2: QKV GEMM (wave w -> head w's q,k,v) ------------
  // mt outer so each xn A-fragment is loaded from LDS once and reused for
  // all three of q/k/v (B-fragments for all 12 tiles preloaded in VGPRs).
  {
    const v16bf* wq = (const v16bf*)wqkv_bf;
    v16bf bf[3][4];
    float bias[3];
#pragma unroll
    for (int which = 0; which < 3; ++which) {
      const int nt = wv + 8 * which;       // N-tile: (which, head wv)
#pragma unroll
      for (int kt = 0; kt < 4; ++kt)
        bf[which][kt] = wq[(nt * 4 + kt) * 32 + lane];
      bias[which] = bqkv[nt * 16 + ln];
    }
#pragma unroll
    for (int mt = 0; mt < 4; ++mt) {
      const int m = mt * 16 + ln;          // A row for this lane
      v16bf a[4];
#pragma unroll
      for (int kt = 0; kt < 4; ++kt) {
        a[kt] = *(const v16bf*)&s_xn[m * 128 + kt * 32 + 16 * h];
      }
#pragma unroll
      for (int which = 0; which < 3; ++which) {
        v8f acc = {};
#pragma unroll
        for (int kt = 0; kt < 4; ++kt)
          acc = wmma_bf16(a[kt], bf[which][kt], acc);
        if (which < 2) {
#pragma unroll
          for (int r = 0; r < 8; ++r) {
            float val = acc[r] + bias[which];
            int row = mt * 16 + r + 8 * h;   // token index
            if (which == 0) s_q[wv][row * 16 + ln] = (__bf16)val;
            else            s_k[wv][row * 16 + ln] = (__bf16)val;
          }
        } else {
          // vT rows are contiguous in r -> one packed 16B store
          v8bf vv;
#pragma unroll
          for (int r = 0; r < 8; ++r) vv[r] = (__bf16)(acc[r] + bias[2]);
          *(v8bf*)&s_vT[wv][ln * 64 + mt * 16 + 8 * h] = vv;
        }
      }
    }
  }
  __syncthreads();  // everyone done with s_xn; reuse it for attn output

  // ---------------- Phase 3: attention (per-wave, head wv) ------------------
  {
    __bf16* attn = s_xn;
    __bf16* slab = s_q[wv];   // probs slab [16][64], reused after q frags load

    // q A-fragments: K = head dim padded 16->32 (high half zero)
    v16bf qa[4];
#pragma unroll
    for (int mt = 0; mt < 4; ++mt) {
      v16bf qv = {};
      if (h == 0) qv = *(const v16bf*)&s_q[wv][(mt * 16 + ln) * 16];
      qa[mt] = qv;
    }
    // k B-fragments: lanes 16..31 carry K=16..31 -> zero padding
    v16bf kb[4];
#pragma unroll
    for (int nt = 0; nt < 4; ++nt) {
      v16bf kv = {};
      if (h == 0) kv = *(const v16bf*)&s_k[wv][(nt * 16 + ln) * 16];
      kb[nt] = kv;
    }
    // v B-fragments for probs@V: K = token (64 -> two chunks of 32)
    v16bf vb[2];
#pragma unroll
    for (int c = 0; c < 2; ++c)
      vb[c] = *(const v16bf*)&s_vT[wv][ln * 64 + 32 * c + 16 * h];

    asm volatile("" ::: "memory");

#pragma unroll
    for (int mt = 0; mt < 4; ++mt) {
      v8f z8 = {};
      v8f t0 = wmma_bf16(qa[mt], kb[0], z8);
      v8f t1 = wmma_bf16(qa[mt], kb[1], z8);
      v8f t2 = wmma_bf16(qa[mt], kb[2], z8);
      v8f t3 = wmma_bf16(qa[mt], kb[3], z8);
      float sv[4][8];
#pragma unroll
      for (int r = 0; r < 8; ++r) {
        sv[0][r] = t0[r] * SCALE_;
        sv[1][r] = t1[r] * SCALE_;
        sv[2][r] = t2[r] * SCALE_;
        // tile nt=3 covers cols 48..63; only col 48 (ln==0) is valid
        sv[3][r] = (ln > 0) ? -1e30f : t3[r] * SCALE_;
      }
      // row-wise softmax (row spread over 16 lanes x 4 tiles)
#pragma unroll
      for (int r = 0; r < 8; ++r) {
        float mx = fmaxf(fmaxf(sv[0][r], sv[1][r]), fmaxf(sv[2][r], sv[3][r]));
#pragma unroll
        for (int o = 1; o < 16; o <<= 1) mx = fmaxf(mx, __shfl_xor(mx, o, 32));
        float e0 = __expf(sv[0][r] - mx);
        float e1 = __expf(sv[1][r] - mx);
        float e2 = __expf(sv[2][r] - mx);
        float e3 = __expf(sv[3][r] - mx);
        float sum = e0 + e1 + e2 + e3;
#pragma unroll
        for (int o = 1; o < 16; o <<= 1) sum += __shfl_xor(sum, o, 32);
        float inv = 1.0f / sum;
        sv[0][r] = e0 * inv;
        sv[1][r] = e1 * inv;
        sv[2][r] = e2 * inv;
        sv[3][r] = e3 * inv;
      }
      // probs -> slab (bf16), cols >=49 are exact zeros -> padding-safe
#pragma unroll
      for (int nt = 0; nt < 4; ++nt)
#pragma unroll
        for (int r = 0; r < 8; ++r)
          slab[(r + 8 * h) * 64 + nt * 16 + ln] = (__bf16)sv[nt][r];
      asm volatile("" ::: "memory");
      // probs A-fragments and probs@V
      v16bf pa[2];
#pragma unroll
      for (int c = 0; c < 2; ++c) {
        pa[c] = *(const v16bf*)&slab[ln * 64 + 32 * c + 16 * h];
      }
      v8f za = {};
      v8f at = wmma_bf16(pa[0], vb[0], za);
      at = wmma_bf16(pa[1], vb[1], at);
#pragma unroll
      for (int r = 0; r < 8; ++r)
        attn[(mt * 16 + r + 8 * h) * 128 + wv * 16 + ln] = (__bf16)at[r];
    }
  }
  __syncthreads();

  // ---------------- Phase 4: output projection + window merge ---------------
  {
    const v16bf* wo = (const v16bf*)wout_bf;
    v16bf bo[4];
#pragma unroll
    for (int kt = 0; kt < 4; ++kt) bo[kt] = wo[(wv * 4 + kt) * 32 + lane];
    const float bias = bout[wv * 16 + ln];
#pragma unroll
    for (int mt = 0; mt < 4; ++mt) {
      const int m = mt * 16 + ln;
      v8f acc = {};
#pragma unroll
      for (int kt = 0; kt < 4; ++kt) {
        v16bf a = *(const v16bf*)&s_xn[m * 128 + kt * 32 + 16 * h];
        acc = wmma_bf16(a, bo[kt], acc);
      }
#pragma unroll
      for (int r = 0; r < 8; ++r) {
        int row = mt * 16 + r + 8 * h;
        if (row < 49) {
          int trow = row / 7, tcol = row - trow * 7;
          int grow = wr * 7 + trow, gcol = wc * 7 + tcol;
          out[((size_t)((b * 56 + grow) * 56 + gcol)) * 128 + wv * 16 + ln] =
              acc[r] + bias;
        }
      }
    }
  }
}

extern "C" void kernel_launch(void* const* d_in, const int* in_sizes, int n_in,
                              void* d_out, int out_size, void* d_ws, size_t ws_size,
                              hipStream_t stream) {
  (void)in_sizes; (void)n_in; (void)out_size; (void)ws_size;
  const float* x     = (const float*)d_in[0];
  const float* gamma = (const float*)d_in[1];
  const float* beta  = (const float*)d_in[2];
  const float* wqkv  = (const float*)d_in[3];
  const float* bqkv  = (const float*)d_in[4];
  const float* wout  = (const float*)d_in[5];
  const float* bout  = (const float*)d_in[6];
  float* out = (float*)d_out;

  __bf16* wsq = (__bf16*)d_ws;                 // 24*4*32*16 bf16 = 96 KB
  __bf16* wso = wsq + 24 * 4 * 32 * 16;        // 8*4*32*16 bf16 = 32 KB

  wconv_kernel<<<(49152 + 255) / 256, 256, 0, stream>>>(wqkv, wsq, 384, 49152);
  wconv_kernel<<<(16384 + 255) / 256, 256, 0, stream>>>(wout, wso, 128, 16384);
  swin_attn_kernel<<<4096, 256, 0, stream>>>(x, gamma, beta, wsq, bqkv, wso,
                                             bout, out);
}
